// MotorCortexRNN_57561151701621
// MI455X (gfx1250) — matrix operand order repaired and checked by reading, not compiled, not run on hardware
//
#include <hip/hip_runtime.h>
#include <cstdint>
#include <cstddef>

// CDNA5 / gfx1250 persistent fused RNN kernel.
// grid = BATCH/16 workgroups, block = 13 waves * 32 = 416 threads.
// Each workgroup owns a 16-row batch tile for all 1000 timesteps.
// Each wave owns a 16-wide slice of the 200 (padded 208) recurrent units:
//   - W fragments (W_rec | W_inp | pad) held in 56 VGPRs for the whole kernel
//   - h slice held in f32 registers (WMMA C/D layout)
//   - h exchanged across waves via a DOUBLE-BUFFERED 16 x 224 f16 LDS buffer
//     -> exactly one workgroup barrier per timestep
//   - all 7 A fragments preloaded per step -> 7 back-to-back v_wmma
//   - x_t software-pipelined one step ahead (wave-uniform duty: waves 0-4)
//   - non-temporal state stores (204.8 MB write-once stream bypasses L2 reuse).

typedef __attribute__((ext_vector_type(16))) _Float16 v16h;
typedef __attribute__((ext_vector_type(8)))  _Float16 v8h;
typedef __attribute__((ext_vector_type(8)))  float    v8f;

#define N_INP    10
#define N_REC    200
#define TSTEPS   1000
#define BATCH    256
#define KPAD     224          // 200 h + 10 x + 14 zero, = 7 * 32
#define NCHUNK   7
#define NWAVES   13           // ceil(208/16) column tiles
#define ALPHA    0.1f
#define NOISE_STD 0.01f

// Deterministic counter-based stand-in for jax.random.normal: one hash chain,
// two 16-bit uniforms -> triangular distribution (zero mean, unit variance).
__device__ __forceinline__ float noise_n01(uint32_t idx) {
    uint32_t a = idx * 0x9E3779B1u;
    a ^= a >> 16; a *= 0x7feb352du; a ^= a >> 15; a *= 0x846ca68bu; a ^= a >> 16;
    float s = (float)(a & 0xFFFFu) + (float)(a >> 16);      // ~ 65535*(u1+u2)
    return s * (2.4494897f / 65535.0f) - 2.4494897f;        // sqrt(6) scaling
}

// Branchless tanh: native V_TANH_F32 when available, else native-exp formula.
#if defined(__gfx1250__) && __has_builtin(__builtin_amdgcn_tanhf)
__device__ __forceinline__ float fast_tanh(float x) {
    return __builtin_amdgcn_tanhf(x);
}
#else
__device__ __forceinline__ float fast_tanh(float x) {
    float xc = fminf(fmaxf(x, -15.0f), 15.0f);
    float e  = __expf(2.0f * xc);
    return (e - 1.0f) * __builtin_amdgcn_rcpf(e + 1.0f);
}
#endif

__global__ __launch_bounds__(NWAVES * 32)
void rnn_persistent_wmma(const float* __restrict__ x,
                         const float* __restrict__ h0,
                         const float* __restrict__ Wi,
                         const float* __restrict__ Wr,
                         const float* __restrict__ br,
                         float* __restrict__ out)
{
    // Double-buffered combined-K h/x exchange: [buf][batch-row m][K] = 14 KB.
    __shared__ _Float16 hbuf[2][16 * KPAD];

    const int tid    = threadIdx.x;
    const int lane   = tid & 31;
    const int wave   = tid >> 5;          // 0..12 : N-tile index
    const int halfhi = lane >> 4;         // 0 = lanes 0-15, 1 = lanes 16-31
    const int l15    = lane & 15;
    const int bBase  = blockIdx.x * 16;   // batch tile origin
    const int mofs   = halfhi * 8;        // C/D layout M offset for this lane
    const int col    = wave * 16 + l15;   // recurrent-unit index (N / B-row)
    const bool colOK = (col < N_REC);

    // Zero both buffers once (cols >= 210 stay zero forever; 0..199 rewritten
    // by h, 200..209 rewritten by x each step).
    for (int i = tid; i < 2 * 16 * KPAD; i += blockDim.x)
        hbuf[0][i] = (_Float16)0.0f;

    // ---- Persistent B fragments: B[k][n] = Wcombined[n][k] ----
    // B k order used here (contiguous, same as A): lanes 0-15 hold K=0..15 of column N=lane,
    // lanes 16-31 hold K=16..31 of column N=lane-16; element e -> K=base+e.
    v16h bfr[NCHUNK];
    #pragma unroll
    for (int c = 0; c < NCHUNK; ++c) {
        const int kb = 32 * c + halfhi * 16;
        #pragma unroll
        for (int e = 0; e < 16; ++e) {
            const int kc = kb + e;
            float wv = 0.0f;
            if (colOK) {
                if (kc < N_REC)               wv = Wr[col * N_REC + kc];
                else if (kc < N_REC + N_INP)  wv = Wi[col * N_INP + (kc - N_REC)];
            }
            bfr[c][e] = (_Float16)wv;
        }
    }
    const float bval = colOK ? br[col] : 0.0f;

    // ---- h slice in f32 registers, WMMA C/D layout:
    // VGPR i, lanes 0-15 -> (M=i, N=l15); lanes 16-31 -> (M=i+8, N=l15).
    float hreg[8];
    #pragma unroll
    for (int i = 0; i < 8; ++i)
        hreg[i] = colOK ? h0[(size_t)(bBase + mofs + i) * N_REC + col] : 0.0f;

    // Per-lane streaming bases (bumped by constant strides each step; the
    // i * TSTEPS * N_REC row offsets fold into 24-bit instruction offsets).
    float* outp = out + ((size_t)(bBase + mofs) * TSTEPS) * N_REC + col;
    uint32_t eb = (uint32_t)(bBase + mofs) * N_REC + (uint32_t)col;

    const int mA = l15;                   // A-fragment batch row this lane owns

    // x duty = waves 0..4 exactly (160 threads = 16 rows x 10 inputs):
    // wave-uniform branch, no exec-mask divergence.
    const bool xduty = (wave < 5);
    const int  xm    = tid / N_INP;
    const int  xi    = tid - xm * N_INP;
    const float* xp  = x + ((size_t)(bBase + xm) * TSTEPS) * N_INP + xi;

    float xv = xduty ? xp[0] : 0.0f;      // pipeline head: x for t=0
    xp += N_INP;

    __syncthreads();

    int p = 0;                            // publish/read buffer for this step
    for (int t = 0; t < TSTEPS; ++t) {
        _Float16* buf = hbuf[p];

        // ---- phase A: publish h (f16) + pipelined x_t into buf[p]
        if (colOK) {
            #pragma unroll
            for (int i = 0; i < 8; ++i)
                buf[(mofs + i) * KPAD + col] = (_Float16)hreg[i];
        }
        if (xduty)
            buf[xm * KPAD + N_REC + xi] = (_Float16)xv;

        // Single barrier per step: also guarantees (via its dscnt-0 wait) that
        // everyone finished reading buf[p] back at step t-2 before we rewrote
        // it above, and that this step's writes are visible below.
        __syncthreads();

        // prefetch x for t+1; consumed after the next barrier, so its latency
        // hides under the WMMA + elementwise phases
        if (xduty && (t + 1 < TSTEPS)) { xv = xp[0]; xp += N_INP; }

        // ---- phase B: preload all A fragments, then 7 chained WMMAs
        // A k order used here (contiguous, same as B): lanes 0-15 K {0..15}+32c,
        //                  lanes 16-31 K {16..31}+32c, row M=l15.
        v16h afr[NCHUNK];
        #pragma unroll
        for (int c = 0; c < NCHUNK; ++c) {
            const int k0 = 32 * c + halfhi * 16;
            const v8h lo = *reinterpret_cast<const v8h*>(&buf[mA * KPAD + k0]);
            const v8h hi = *reinterpret_cast<const v8h*>(&buf[mA * KPAD + k0 + 8]);
            #pragma unroll
            for (int e = 0; e < 8; ++e) { afr[c][e] = lo[e]; afr[c][8 + e] = hi[e]; }
        }

        v8f acc;
        #pragma unroll
        for (int i = 0; i < 8; ++i) acc[i] = bval;
        #pragma unroll
        for (int c = 0; c < NCHUNK; ++c) {
            acc = __builtin_amdgcn_wmma_f32_16x16x32_f16(
                      /*neg_a=*/false, afr[c], /*neg_b=*/false, bfr[c],
                      /*c_mod=*/(short)0, acc, /*reuse_a=*/false, /*reuse_b=*/false);
        }

        // ---- phase C: branchless elementwise update ...
        #pragma unroll
        for (int i = 0; i < 8; ++i) {
            const float rate = fast_tanh(acc[i]);
            const float eta  = NOISE_STD * noise_n01(eb + (uint32_t)(i * N_REC));
            float h = hreg[i];
            hreg[i] = h + ALPHA * (-h + rate + eta);
        }
        // ... then one exec-masked block of 8 non-temporal streaming stores
        if (colOK) {
            #pragma unroll
            for (int i = 0; i < 8; ++i)
                __builtin_nontemporal_store(hreg[i], &outp[(size_t)i * (TSTEPS * N_REC)]);
        }
        outp += N_REC;                       // advance to timestep t+1
        eb   += (uint32_t)BATCH * N_REC;
        p ^= 1;
    }
}

extern "C" void kernel_launch(void* const* d_in, const int* in_sizes, int n_in,
                              void* d_out, int out_size, void* d_ws, size_t ws_size,
                              hipStream_t stream) {
    (void)in_sizes; (void)n_in; (void)out_size; (void)d_ws; (void)ws_size;
    const float* x  = (const float*)d_in[0];   // [256,1000,10]
    const float* h0 = (const float*)d_in[1];   // [256,200]
    const float* Wi = (const float*)d_in[2];   // [200,10]
    const float* Wr = (const float*)d_in[3];   // [200,200]
    const float* br = (const float*)d_in[4];   // [200]
    float* out = (float*)d_out;                // [256,1000,200]

    dim3 grid(BATCH / 16);
    dim3 block(NWAVES * 32);
    rnn_persistent_wmma<<<grid, block, 0, stream>>>(x, h0, Wi, Wr, br, out);
}
